// DRQN_15719580304521
// MI455X (gfx1250) — matrix operand order repaired and checked by reading, not compiled, not run on hardware
//
#include <hip/hip_runtime.h>
#include <hip/hip_bf16.h>
#include <stdint.h>

// ---------------------------------------------------------------------------
// DRQN on MI455X (gfx1250): bf16 WMMA GEMMs for all dense layers.
//   Phase 1 (parallel over T*B=131072 rows):
//     A1 = relu(Xbf @ W1a^T + b1a)          [131072 x 512]
//     A  = relu(A1  @ W1b^T + b1b)          [131072 x 256]
//     GI = A @ W_ih^T + b_ih  (fp32)        [131072 x 768]
//   Phase 2 (sequential, 512 steps):
//     GH = h_bf @ W_hh^T + b_hh  (256x768 WMMA GEMM)
//     gates (elementwise) -> h_new ; store h_new as bf16 into Hbf[t]
//   Phase 3 (parallel):
//     A2  = relu(Hbf @ W2a^T + b2a)         [131072 x 512]
//     out = A2 @ W2b^T + b2b                [131072 x 18] (N padded to 32)
//
// GEMM kernel: one wave computes a 16 x (NT*16) tile, K fully unrolled
// (template K) with an explicit depth-2 software pipeline. A
// __builtin_amdgcn_sched_barrier(0) between the prefetch loads (chunk ci+2)
// and the WMMA group (chunk ci) stops the machine scheduler from sinking
// loads back to their consumers, preserving the prefetch distance so waits
// become partial instead of full s_wait_loadcnt 0x0 drains.
// ---------------------------------------------------------------------------

typedef __attribute__((ext_vector_type(16))) __bf16 v16bf;
typedef __attribute__((ext_vector_type(8)))  __bf16 v8bf;
typedef __attribute__((ext_vector_type(8)))  float  v8f;

static __device__ __forceinline__ __bf16 f2bf(float f) {
    union { float f; uint32_t u; } v; v.f = f;
    uint32_t r = v.u + 0x7FFFu + ((v.u >> 16) & 1u);   // round-to-nearest-even
    return __builtin_bit_cast(__bf16, (uint16_t)(r >> 16));
}

// two contiguous 16B runs -> one 16-element bf16 fragment
static __device__ __forceinline__ v16bf frag2(const __bf16* p0, const __bf16* p1) {
    v8bf lo = *reinterpret_cast<const v8bf*>(p0);
    v8bf hi = *reinterpret_cast<const v8bf*>(p1);
    v16bf f;
#pragma unroll
    for (int i = 0; i < 8; ++i) { f[i] = lo[i]; f[i + 8] = hi[i]; }
    return f;
}

// ------------------------------- converts ----------------------------------
__global__ void k_f32_to_bf16(const float* __restrict__ src,
                              __bf16* __restrict__ dst, int n) {
    int i = blockIdx.x * blockDim.x + threadIdx.x;
    if (i < n) dst[i] = f2bf(src[i]);
}

// convert [rows_src, cols] fp32 -> [rows_dst, cols] bf16 with zero row padding
__global__ void k_f32_to_bf16_padrows(const float* __restrict__ src,
                                      __bf16* __restrict__ dst,
                                      int rows_src, int rows_dst, int cols) {
    int i = blockIdx.x * blockDim.x + threadIdx.x;
    if (i >= rows_dst * cols) return;
    int r = i / cols, c = i - r * cols;
    dst[i] = (r < rows_src) ? f2bf(src[(size_t)r * cols + c])
                            : __builtin_bit_cast(__bf16, (uint16_t)0);
}

__global__ void k_zero_h(float* __restrict__ hf, __bf16* __restrict__ hb, int n) {
    int i = blockIdx.x * blockDim.x + threadIdx.x;
    if (i < n) { hf[i] = 0.0f; hb[i] = __builtin_bit_cast(__bf16, (uint16_t)0); }
}

// ------------------------------- WMMA GEMM ---------------------------------
// C[M,Nreal] = act(A[M,K](bf16) * B[N,K]^T(bf16) + bias[Nreal])
// One wave32 computes a 16 x (NT*16) tile via V_WMMA_F32_16X16X32_BF16.
// Fragment layouts per CDNA5 ISA 7.12.2:
//   A 16x32 bf16: lane = {m = lane&15, kh = lane>>4}; elems 0..7 -> k=kh*8+e,
//                 elems 8..15 -> k=16+kh*8+(e-8)  (two contiguous 16B runs)
//   B 32x16 bf16: lane = {n = lane&15, kh}; same k order as A (two 16B runs)
//   C 16x16 f32 : VGPR v -> m = v + 8*kh, n = lane&15
template <int NT, int K, bool RELU, bool BF16OUT>
__global__ __launch_bounds__(256)
void k_gemm_bf16(const __bf16* __restrict__ A, const __bf16* __restrict__ B,
                 const float* __restrict__ bias, void* __restrict__ Cout,
                 int Nreal, int tilesN /* in NT*16-col groups */,
                 int totalTiles) {
    const int wave = threadIdx.x >> 5;
    const int lane = threadIdx.x & 31;
    const int tile = blockIdx.x * 8 + wave;
    if (tile >= totalTiles) return;            // whole-wave early exit (EXEC stays full)
    const int tm = tile / tilesN;
    const int tn = tile - tm * tilesN;
    const int m0 = tm << 4;
    const int n0 = tn * (NT << 4);
    const int lm = lane & 15;
    const int kh = lane >> 4;

    const __bf16* __restrict__ Ap = A + (size_t)(m0 + lm) * K + kh * 8;
    const __bf16* __restrict__ Bp = B + (size_t)(n0 + lm) * K + kh * 8;

    struct Frags { v16bf a; v16bf b[NT]; };
    auto loadChunk = [&](int ci) -> Frags {
        const int k0 = ci * 32;
        Frags f;
        f.a = frag2(Ap + k0, Ap + k0 + 16);
#pragma unroll
        for (int j = 0; j < NT; ++j) {
            const __bf16* Bpj = Bp + (size_t)(j << 4) * K + k0;
            f.b[j] = frag2(Bpj, Bpj + 16);
        }
        return f;
    };

    v8f c[NT];
#pragma unroll
    for (int j = 0; j < NT; ++j) c[j] = v8f{};

    constexpr int NCH = K / 32;                // K-chunks
    constexpr int PF  = (NCH >= 2) ? 2 : 1;    // software-pipeline depth

    Frags buf[PF];
#pragma unroll
    for (int i = 0; i < PF; ++i) buf[i] = loadChunk(i);

#pragma unroll
    for (int ci = 0; ci < NCH; ++ci) {
        Frags cur = buf[ci % PF];              // copy out (SSA-renamed after unroll)
        if (ci + PF < NCH) buf[ci % PF] = loadChunk(ci + PF);  // prefetch ahead
        // Fence: prefetch loads must be ISSUED before this chunk's WMMAs in
        // the final schedule (keeps ~2 chunks of loads in flight -> partial
        // s_wait_loadcnt instead of full drains).
        __builtin_amdgcn_sched_barrier(0);
#pragma unroll
        for (int j = 0; j < NT; ++j) {
            c[j] = __builtin_amdgcn_wmma_f32_16x16x32_bf16(
                       /*neg_a=*/false, cur.a, /*neg_b=*/false, cur.b[j],
                       /*c_mod=*/(short)0, c[j],
                       /*reuse_a=*/false, /*reuse_b=*/false);
        }
        __builtin_amdgcn_sched_barrier(0);
    }

#pragma unroll
    for (int j = 0; j < NT; ++j) {
        const int n = n0 + (j << 4) + lm;
        const float bv = (n < Nreal) ? bias[n] : 0.0f;
#pragma unroll
        for (int v = 0; v < 8; ++v) {
            const int m = m0 + v + (kh << 3);
            float val = c[j][v] + bv;
            if (RELU) val = val > 0.0f ? val : 0.0f;
            if (n < Nreal) {
                if (BF16OUT) ((__bf16*)Cout)[(size_t)m * Nreal + n] = f2bf(val);
                else         ((float*)Cout)[(size_t)m * Nreal + n] = val;
            }
        }
    }
}

// ------------------------------- GRU gates ---------------------------------
// B=256, H=256, 3H=768 hardcoded.
__global__ __launch_bounds__(256)
void k_gru_gate(const float* __restrict__ GIt, const float* __restrict__ GH,
                float* __restrict__ hf, __bf16* __restrict__ hb,
                __bf16* __restrict__ Hbf_t) {
    const int idx = blockIdx.x * blockDim.x + threadIdx.x;   // < 65536
    const int b = idx >> 8;
    const int hid = idx & 255;
    const float* gi = GIt + (size_t)b * 768;
    const float* gh = GH  + (size_t)b * 768;
    const float ir = gi[hid], iz = gi[256 + hid], in_ = gi[512 + hid];
    const float hr = gh[hid], hz = gh[256 + hid], hn = gh[512 + hid];
    const float r = 1.0f / (1.0f + __expf(-(ir + hr)));
    const float z = 1.0f / (1.0f + __expf(-(iz + hz)));
    const float n = tanhf(in_ + r * hn);
    const float hprev = hf[idx];
    const float hnew = (1.0f - z) * n + z * hprev;
    hf[idx] = hnew;
    const __bf16 hbf = f2bf(hnew);
    hb[idx] = hbf;
    Hbf_t[idx] = hbf;
}

// ------------------------------- launcher ----------------------------------
extern "C" void kernel_launch(void* const* d_in, const int* in_sizes, int n_in,
                              void* d_out, int out_size, void* d_ws, size_t ws_size,
                              hipStream_t stream) {
    (void)in_sizes; (void)n_in; (void)out_size; (void)ws_size;
    const int T = 512, B = 256, IN = 256, H1 = 512, H = 256, OUT = 18;
    const int R = T * B;                     // 131072 rows
    const int G3 = 3 * H;                    // 768

    const float* x   = (const float*)d_in[0];
    const float* W1a = (const float*)d_in[1];
    const float* b1a = (const float*)d_in[2];
    const float* W1b = (const float*)d_in[3];
    const float* b1b = (const float*)d_in[4];
    const float* Wih = (const float*)d_in[5];
    const float* bih = (const float*)d_in[6];
    const float* Whh = (const float*)d_in[7];
    const float* bhh = (const float*)d_in[8];
    const float* W2a = (const float*)d_in[9];
    const float* b2a = (const float*)d_in[10];
    const float* W2b = (const float*)d_in[11];
    const float* b2b = (const float*)d_in[12];
    float* out = (float*)d_out;

    // --- workspace bump allocator (256B aligned) ---
    char* ws = (char*)d_ws;
    size_t off = 0;
    auto alloc = [&](size_t bytes) -> char* {
        char* p = ws + off;
        off += (bytes + 255) & ~(size_t)255;
        return p;
    };
    __bf16* Xbf  = (__bf16*)alloc((size_t)R * IN * 2);   // slot also reused as Hbf
    __bf16* A1   = (__bf16*)alloc((size_t)R * H1 * 2);   // slot also reused as A2
    __bf16* Abf  = (__bf16*)alloc((size_t)R * H  * 2);
    float*  GI   = (float*) alloc((size_t)R * G3 * 4);
    float*  GH   = (float*) alloc((size_t)B * G3 * 4);
    float*  hf   = (float*) alloc((size_t)B * H * 4);
    __bf16* hb   = (__bf16*)alloc((size_t)B * H * 2);
    __bf16* W1ab = (__bf16*)alloc((size_t)H1 * IN * 2);
    __bf16* W1bb = (__bf16*)alloc((size_t)H  * H1 * 2);
    __bf16* Wihb = (__bf16*)alloc((size_t)G3 * H  * 2);
    __bf16* Whhb = (__bf16*)alloc((size_t)G3 * H  * 2);
    __bf16* W2ab = (__bf16*)alloc((size_t)H1 * H  * 2);
    __bf16* W2bb = (__bf16*)alloc((size_t)32 * H1 * 2);  // W_fc2b padded 18->32 rows
    // aliases (live ranges do not overlap):
    __bf16* Hbf = Xbf;   // Xbf dead after GEMM1; Hbf live from GRU on
    __bf16* A2  = A1;    // A1 dead after GEMM2; A2 live in fc2 phase

    auto cvblocks = [](int n) { return (n + 255) / 256; };

    // --- convert inputs / weights to bf16 ---
    k_f32_to_bf16<<<cvblocks(R * IN), 256, 0, stream>>>(x, Xbf, R * IN);
    k_f32_to_bf16<<<cvblocks(H1 * IN), 256, 0, stream>>>(W1a, W1ab, H1 * IN);
    k_f32_to_bf16<<<cvblocks(H * H1), 256, 0, stream>>>(W1b, W1bb, H * H1);
    k_f32_to_bf16<<<cvblocks(G3 * H), 256, 0, stream>>>(Wih, Wihb, G3 * H);
    k_f32_to_bf16<<<cvblocks(G3 * H), 256, 0, stream>>>(Whh, Whhb, G3 * H);
    k_f32_to_bf16<<<cvblocks(H1 * H), 256, 0, stream>>>(W2a, W2ab, H1 * H);
    k_f32_to_bf16_padrows<<<cvblocks(32 * H1), 256, 0, stream>>>(W2b, W2bb, OUT, 32, H1);

    auto gemm_blocks = [](int tiles) { return (tiles + 7) / 8; };

    // --- phase 1: time-parallel fc1 + input gate projection ---
    {   // A1 = relu(Xbf @ W1a^T + b1a)   [R x 512], K=256, 16x64 tiles
        int tn = H1 / 64, tot = (R / 16) * tn;
        k_gemm_bf16<4, 256, true, true><<<gemm_blocks(tot), 256, 0, stream>>>(
            Xbf, W1ab, b1a, A1, H1, tn, tot);
    }
    {   // Abf = relu(A1 @ W1b^T + b1b)   [R x 256], K=512
        int tn = H / 64, tot = (R / 16) * tn;
        k_gemm_bf16<4, 512, true, true><<<gemm_blocks(tot), 256, 0, stream>>>(
            A1, W1bb, b1b, Abf, H, tn, tot);
    }
    {   // GI = Abf @ Wih^T + bih (fp32)  [R x 768], K=256
        int tn = G3 / 64, tot = (R / 16) * tn;
        k_gemm_bf16<4, 256, false, false><<<gemm_blocks(tot), 256, 0, stream>>>(
            Abf, Wihb, bih, GI, G3, tn, tot);
    }

    // --- phase 2: sequential GRU ---
    k_zero_h<<<cvblocks(B * H), 256, 0, stream>>>(hf, hb, B * H);
    {
        const int tn = G3 / 64;                 // 12 column-groups
        const int tot = (B / 16) * tn;          // 192 tiles -> 24 blocks
        for (int t = 0; t < T; ++t) {
            k_gemm_bf16<4, 256, false, false><<<gemm_blocks(tot), 256, 0, stream>>>(
                hb, Whhb, bhh, GH, G3, tn, tot);
            k_gru_gate<<<(B * H) / 256, 256, 0, stream>>>(
                GI + (size_t)t * B * G3, GH, hf, hb, Hbf + (size_t)t * B * H);
        }
    }

    // --- phase 3: time-parallel fc2 head ---
    {   // A2 = relu(Hbf @ W2a^T + b2a)   [R x 512], K=256
        int tn = H1 / 64, tot = (R / 16) * tn;
        k_gemm_bf16<4, 256, true, true><<<gemm_blocks(tot), 256, 0, stream>>>(
            Hbf, W2ab, b2a, A2, H1, tn, tot);
    }
    {   // out = A2 @ W2b^T + b2b (fp32)  [R x 18], N padded to 32 (NT=2), K=512
        int tn = 1, tot = (R / 16) * tn;
        k_gemm_bf16<2, 512, false, false><<<gemm_blocks(tot), 256, 0, stream>>>(
            A2, W2bb, b2b, out, OUT, tn, tot);
    }
}
